// Gemma4Attention_16724602650966
// MI455X (gfx1250) — compile-verified, matrix operand order checked by reading, not run on hardware
//
#include <hip/hip_runtime.h>
#include <stdint.h>

// ---------------------------------------------------------------------------
// Gemma-style attention block for MI455X (gfx1250), wave32 + WMMA bf16.
//   B=2, S=2048, E=2048, NH=8, NKV=4, HD=256, ROT=128, WIN=1024, CAP=50
// ---------------------------------------------------------------------------

#define BB 2
#define SS 2048
#define EE 2048
#define NH 8
#define NKV 4
#define HD 256
#define ROT 128
#define WIN 1024
#define BS (BB * SS)          // 4096 rows in all GEMMs

#define CAP 50.0f
#define INV_CAP 0.02f
#define ATT_SCALE 0.0625f     // 256^-0.5
#define NEG_BIG (-2.3819763e38f)
#define M_INIT  (-1.0e30f)
#define EPS 1e-6f
#define ROPE_BASE 10000.0f

typedef __bf16  bf16x16 __attribute__((ext_vector_type(16)));
typedef float   floatx8 __attribute__((ext_vector_type(8)));
typedef int     v4i     __attribute__((vector_size(16)));   // matches builtin param

union FragAB { bf16x16 v; uint32_t u[8]; uint4 q[2]; };

// gfx1250 async global->LDS copy (ASYNCcnt path); builtin existence confirmed.
#if defined(__has_builtin)
#if __has_builtin(__builtin_amdgcn_global_load_async_to_lds_b128)
#define HAVE_ASYNC_LDS 1
#endif
#endif
#ifndef HAVE_ASYNC_LDS
#define HAVE_ASYNC_LDS 0
#endif

typedef __attribute__((address_space(1))) v4i* gptr_v4i;
typedef __attribute__((address_space(3))) v4i* lptr_v4i;

__device__ __forceinline__ void g2l_b128(const void* g, void* l) {
#if HAVE_ASYNC_LDS
    __builtin_amdgcn_global_load_async_to_lds_b128(
        (gptr_v4i)(uintptr_t)g, (lptr_v4i)(uintptr_t)l, 0, 0);
#else
    *(uint4*)l = *(const uint4*)g;
#endif
}

__device__ __forceinline__ void wait_g2l() {
#if HAVE_ASYNC_LDS
#if defined(__has_builtin) && __has_builtin(__builtin_amdgcn_s_wait_asynccnt)
    __builtin_amdgcn_s_wait_asynccnt(0);
#else
    asm volatile("s_wait_asynccnt 0" ::: "memory");
#endif
#endif
}

// pack two f32 -> packed bf16 dword (compiler emits v_cvt_pk_bf16_f32)
__device__ __forceinline__ uint32_t pkbf(float a, float b) {
    union { __bf16 h[2]; uint32_t u; } p;
    p.h[0] = (__bf16)a; p.h[1] = (__bf16)b;
    return p.u;
}

// 16-bit A-matrix (16x32) fragment, per CDNA5 ISA 7.12.2: for lane L the 16
// elements are two contiguous 16B runs of the lane's row:
//   bytes [hi*16 .. hi*16+15]  (K = hi*8 .. hi*8+7)
//   bytes [32+hi*16 .. +15]    (K = 16+hi*8 .. 16+hi*8+7),  hi = (L>=16)
// -> exactly two ds_load_b128 per fragment.
__device__ __forceinline__ FragAB load_frag(const __bf16* rowbase, int hi) {
    FragAB f;
    f.q[0] = *(const uint4*)(rowbase + hi * 8);
    f.q[1] = *(const uint4*)(rowbase + 16 + hi * 8);
    return f;
}

__device__ __forceinline__ float rowmax16(float x) {
    x = fmaxf(x, __shfl_xor(x, 1, 32));
    x = fmaxf(x, __shfl_xor(x, 2, 32));
    x = fmaxf(x, __shfl_xor(x, 4, 32));
    x = fmaxf(x, __shfl_xor(x, 8, 32));
    return x;
}
__device__ __forceinline__ float rowsum16(float x) {
    x += __shfl_xor(x, 1, 32);
    x += __shfl_xor(x, 2, 32);
    x += __shfl_xor(x, 4, 32);
    x += __shfl_xor(x, 8, 32);
    return x;
}

// ---------------------------------------------------------------------------
// GEMM: C[M x N] = A[M x K] * B[K x N], fp32 inputs, bf16 WMMA, OutT output.
// Block = 256 threads (8 waves), 64x64 tile; wave -> 16x32 via two WMMA tiles.
// ---------------------------------------------------------------------------
template <typename OutT>
__global__ void __launch_bounds__(256)
gemm_bf16_wmma_kernel(const float* __restrict__ A, const float* __restrict__ Bm,
                      OutT* __restrict__ C, int N, int K) {
    __shared__ __align__(16) __bf16 As[64 * 32];   // [m][k]
    __shared__ __align__(16) __bf16 Bs[64 * 32];   // [n][k] (K-pairs contiguous)

    const int tid  = threadIdx.x;
    const int lane = tid & 31;
    const int wv   = tid >> 5;
    const int hi   = (lane >= 16) ? 1 : 0;
    const int m0   = blockIdx.y * 64;
    const int n0   = blockIdx.x * 64;
    const int mt   = (wv >> 1) * 16;
    const int nt   = (wv & 1) * 32;

    floatx8 acc0, acc1;
#pragma unroll
    for (int r = 0; r < 8; ++r) { acc0[r] = 0.0f; acc1[r] = 0.0f; }

    for (int k0 = 0; k0 < K; k0 += 32) {
        // ---- phase 1: batched global loads into registers ----
        float4 av[2];
#pragma unroll
        for (int i = 0; i < 2; ++i) {
            int e4 = tid + 256 * i;                 // 512 float4's = 64x32 tile
            int m = e4 >> 3, kf = (e4 & 7) * 4;
            av[i] = *(const float4*)&A[(size_t)(m0 + m) * K + (k0 + kf)];
        }
        const int bn = tid & 63;
        const int kbase = (tid >> 6) * 8;
        float bv[8];
#pragma unroll
        for (int i = 0; i < 8; ++i)
            bv[i] = Bm[(size_t)(k0 + kbase + i) * N + (n0 + bn)];

        if (k0 + 32 < K)
            __builtin_prefetch(&A[(size_t)(m0 + (tid >> 2)) * K + k0 + 32 + ((tid & 3) << 3)], 0, 1);

        // ---- phase 2: pack + wide LDS stores ----
#pragma unroll
        for (int i = 0; i < 2; ++i) {
            int e4 = tid + 256 * i;
            int m = e4 >> 3, kf = (e4 & 7) * 4;
            uint2 t2; t2.x = pkbf(av[i].x, av[i].y); t2.y = pkbf(av[i].z, av[i].w);
            *(uint2*)&As[m * 32 + kf] = t2;          // ds_store_b64
        }
#pragma unroll
        for (int i = 0; i < 4; ++i)
            *(uint32_t*)&Bs[bn * 32 + kbase + 2 * i] = pkbf(bv[2 * i], bv[2 * i + 1]);
        __syncthreads();

        FragAB a  = load_frag(&As[(mt + (lane & 15)) * 32], hi);
        FragAB b0 = load_frag(&Bs[(nt + (lane & 15)) * 32], hi);
        FragAB b1 = load_frag(&Bs[(nt + 16 + (lane & 15)) * 32], hi);
        acc0 = __builtin_amdgcn_wmma_f32_16x16x32_bf16(false, a.v, false, b0.v,
                                                       (short)0, acc0, false, false);
        acc1 = __builtin_amdgcn_wmma_f32_16x16x32_bf16(false, a.v, false, b1.v,
                                                       (short)0, acc1, false, false);
        __syncthreads();
    }

    // C/D layout: VGPR r of lane L -> (M = r + 8*(L>=16), N = L&15)
#pragma unroll
    for (int r = 0; r < 8; ++r) {
        size_t row = (size_t)(m0 + mt + r + 8 * hi);
        C[row * N + (n0 + nt + (lane & 15))]      = (OutT)acc0[r];
        C[row * N + (n0 + nt + 16 + (lane & 15))] = (OutT)acc1[r];
    }
}

// ---------------------------------------------------------------------------
// Per-head RMSNorm + partial interleaved RoPE + scale; fp32 in, bf16 out.
// One block of 256 threads per (b, s, head).
// ---------------------------------------------------------------------------
__global__ void __launch_bounds__(256)
rmsnorm_rope_kernel(const float* __restrict__ in, __bf16* __restrict__ out,
                    const float* __restrict__ scale, int nheads, float outscale) {
    const int tid = threadIdx.x;
    const long long blk = blockIdx.x;
    const int h = (int)(blk % nheads);
    const long long bs = blk / nheads;
    const int pos = (int)(bs % SS);
    const float* vin = in + (size_t)bs * nheads * HD + (size_t)h * HD;
    __bf16* vout = out + (size_t)bs * nheads * HD + (size_t)h * HD;

    float x = vin[tid];
    float ss = x * x;
#pragma unroll
    for (int m = 1; m < 32; m <<= 1) ss += __shfl_xor(ss, m, 32);

    __shared__ float wsum[8];
    __shared__ float s_inv;
    __shared__ float nrm[HD];
    if ((tid & 31) == 0) wsum[tid >> 5] = ss;
    __syncthreads();
    if (tid == 0) {
        float tt = 0.0f;
#pragma unroll
        for (int i = 0; i < 8; ++i) tt += wsum[i];
        s_inv = rsqrtf(tt * (1.0f / HD) + EPS);
    }
    __syncthreads();

    float n = x * s_inv * scale[tid];
    nrm[tid] = n;
    __syncthreads();

    if (tid < 64) {                       // RoPE on d in [0,128), interleaved pairs
        float theta = powf(ROPE_BASE, -(float)tid * (1.0f / 64.0f));
        float ang = theta * (float)pos;
        float c = cosf(ang), sn = sinf(ang);
        float x0 = nrm[2 * tid], x1 = nrm[2 * tid + 1];
        vout[2 * tid]     = (__bf16)((x0 * c - x1 * sn) * outscale);
        vout[2 * tid + 1] = (__bf16)((x0 * sn + x1 * c) * outscale);
    } else if (tid >= 128) {              // pass-through half of head dim
        vout[tid] = (__bf16)(n * outscale);
    }
}

// ---------------------------------------------------------------------------
// Flash attention, sliding window + logit softcap, bf16 WMMA, bf16 inputs.
// grid = (S/128, NH, B); block = 256 (8 waves). Wave w -> 16 query rows.
// Q/K tiles staged via async global->LDS (b128/lane); V transposed manually.
// ---------------------------------------------------------------------------
#define QS_ELEMS (8 * 16 * 256)
#define KS_ELEMS (32 * 256)
#define VT_ELEMS (256 * 32)
#define PS_ELEMS (8 * 16 * 32)
#define FLASH_LDS_BYTES ((QS_ELEMS + KS_ELEMS + VT_ELEMS + PS_ELEMS) * 2)

__global__ void __launch_bounds__(256)
flash_attn_kernel(const __bf16* __restrict__ q, const __bf16* __restrict__ k,
                  const __bf16* __restrict__ v, float* __restrict__ o) {
    extern __shared__ char smem[];
    __bf16* Qs = (__bf16*)smem;            // [128 rows][256 d]
    __bf16* Ks = Qs + QS_ELEMS;            // [32 keys][256 d]
    __bf16* Vt = Ks + KS_ELEMS;            // [256 d][32 keys]
    __bf16* Ps = Vt + VT_ELEMS;            // [8][16][32]

    const int tid = threadIdx.x;
    const int lane = tid & 31;
    const int wv = tid >> 5;
    const int hi = (lane >= 16) ? 1 : 0;

    const int qblock = blockIdx.x;         // 128-row query block
    const int h = blockIdx.y;
    const int b = blockIdx.z;
    const int hk = h >> 1;                 // GQA: NH/NKV = 2

    const int qb  = qblock * 128 + wv * 16;
    const int jlo = qb - (WIN - 1);
    const int jhi = qb + 15;

    // Stage this block's Q tile (128 rows x 256 bf16 = 64KB), 16B per lane-op.
    {
        const char* qbase = (const char*)q;
#pragma unroll
        for (int i = 0; i < 16; ++i) {
            int off = (tid + 256 * i) * 16;          // byte offset in Qs
            int row = off >> 9, db = off & 511;      // 512 B per row
            const char* g = qbase +
                ((size_t)(b * SS + qblock * 128 + row) * (NH * HD) + h * HD) * 2 + db;
            g2l_b128(g, (char*)Qs + off);
        }
    }
    wait_g2l();

    floatx8 acc[16];
    float m_r[8], l_r[8];
#pragma unroll
    for (int n2 = 0; n2 < 16; ++n2)
#pragma unroll
        for (int r = 0; r < 8; ++r) acc[n2][r] = 0.0f;
#pragma unroll
    for (int r = 0; r < 8; ++r) { m_r[r] = M_INIT; l_r[r] = 0.0f; }

    int kstart = qblock * 128 - (WIN - 1);
    if (kstart < 0) kstart = 0;
    kstart &= ~31;
    const int kend = qblock * 128 + 127;

    for (int kb = kstart; kb <= kend; kb += 32) {
        __syncthreads();                   // previous compute done before restage
        // K tile [32 keys][256 d]: contiguous 512B rows -> async b128 copies.
        {
            const char* kbase = (const char*)k;
#pragma unroll
            for (int i = 0; i < 4; ++i) {
                int off = (tid + 256 * i) * 16;
                int row = off >> 9, db = off & 511;
                const char* g = kbase +
                    ((size_t)(b * SS + kb + row) * (NKV * HD) + hk * HD) * 2 + db;
                g2l_b128(g, (char*)Ks + off);
            }
        }
        // V transposed tile [256 d][32 keys]: batched bf16 loads, dword stores.
        {
            const int d = tid;             // one head-dim column per thread
            __bf16 vr[32];
#pragma unroll
            for (int i = 0; i < 32; ++i)
                vr[i] = v[(size_t)(b * SS + kb + i) * (NKV * HD) + hk * HD + d];
#pragma unroll
            for (int i = 0; i < 16; ++i) {
                union { __bf16 h2[2]; uint32_t u; } p;
                p.h2[0] = vr[2 * i]; p.h2[1] = vr[2 * i + 1];
                *(uint32_t*)&Vt[d * 32 + 2 * i] = p.u;
            }
        }
        wait_g2l();
        __syncthreads();

        const bool active = !(kb > jhi || (kb + 31) < jlo);
        if (active) {
            // ---- scores: two 16x16 tiles over 32 keys, contraction = 256 ----
            const __bf16* qrow  = &Qs[(wv * 16 + (lane & 15)) * 256];
            const __bf16* k0row = &Ks[(lane & 15) * 256];
            const __bf16* k1row = &Ks[(16 + (lane & 15)) * 256];
            floatx8 c0, c1;
#pragma unroll
            for (int r = 0; r < 8; ++r) { c0[r] = 0.0f; c1[r] = 0.0f; }
#pragma unroll
            for (int d0 = 0; d0 < 256; d0 += 32) {
                FragAB a  = load_frag(qrow + d0, hi);
                FragAB b0 = load_frag(k0row + d0, hi);
                FragAB b1 = load_frag(k1row + d0, hi);
                c0 = __builtin_amdgcn_wmma_f32_16x16x32_bf16(false, a.v, false, b0.v,
                                                             (short)0, c0, false, false);
                c1 = __builtin_amdgcn_wmma_f32_16x16x32_bf16(false, a.v, false, b1.v,
                                                             (short)0, c1, false, false);
            }

            // ---- softcap + mask + online softmax ----
            float alpha[8];
#pragma unroll
            for (int r = 0; r < 8; ++r) {
                const int i  = qb + r + 8 * hi;
                const int j0 = kb + (lane & 15);
                const int j1 = j0 + 16;
                float s0 = tanhf(c0[r] * INV_CAP) * CAP;
                float s1 = tanhf(c1[r] * INV_CAP) * CAP;
                if (!((j0 <= i) && ((i - j0) < WIN))) s0 = NEG_BIG;
                if (!((j1 <= i) && ((i - j1) < WIN))) s1 = NEG_BIG;
                float rm = rowmax16(fmaxf(s0, s1));
                float mn = fmaxf(m_r[r], rm);           // floor at M_INIT
                float al = __expf(m_r[r] - mn);
                float p0 = __expf(s0 - mn);             // masked -> exp(-huge)=0
                float p1 = __expf(s1 - mn);
                l_r[r] = l_r[r] * al + rowsum16(p0 + p1);
                m_r[r] = mn;
                alpha[r] = al;
                Ps[(wv * 16 + r + 8 * hi) * 32 + (lane & 15)]      = (__bf16)p0;
                Ps[(wv * 16 + r + 8 * hi) * 32 + 16 + (lane & 15)] = (__bf16)p1;
            }
#pragma unroll
            for (int n2 = 0; n2 < 16; ++n2)
#pragma unroll
                for (int r = 0; r < 8; ++r) acc[n2][r] *= alpha[r];

            // ---- P (C-layout -> A-fragment via LDS) then P * V ----
            FragAB pa = load_frag(&Ps[(wv * 16 + (lane & 15)) * 32], hi);
#pragma unroll
            for (int n2 = 0; n2 < 16; ++n2) {
                FragAB vb = load_frag(&Vt[(n2 * 16 + (lane & 15)) * 32], hi);
                acc[n2] = __builtin_amdgcn_wmma_f32_16x16x32_bf16(false, pa.v, false, vb.v,
                                                                  (short)0, acc[n2], false, false);
            }
        }
    }

    // ---- epilogue: divide by l, store fp32 [b, s, h*HD + d] ----
    float rl[8];
#pragma unroll
    for (int r = 0; r < 8; ++r) rl[r] = 1.0f / fmaxf(l_r[r], 1e-20f);
#pragma unroll
    for (int n2 = 0; n2 < 16; ++n2)
#pragma unroll
        for (int r = 0; r < 8; ++r) {
            int i = qb + r + 8 * hi;
            int d = n2 * 16 + (lane & 15);
            o[(size_t)(b * SS + i) * (NH * HD) + h * HD + d] = acc[n2][r] * rl[r];
        }
}

// ---------------------------------------------------------------------------
// Host-side launch
// ---------------------------------------------------------------------------
extern "C" void kernel_launch(void* const* d_in, const int* in_sizes, int n_in,
                              void* d_out, int out_size, void* d_ws, size_t ws_size,
                              hipStream_t stream) {
    (void)in_sizes; (void)n_in; (void)out_size; (void)ws_size;

    const float* x       = (const float*)d_in[0];
    const float* Wq      = (const float*)d_in[1];
    const float* Wk      = (const float*)d_in[2];
    const float* Wv      = (const float*)d_in[3];
    const float* Wo      = (const float*)d_in[4];
    const float* q_scale = (const float*)d_in[5];
    const float* k_scale = (const float*)d_in[6];

    float*  q_ws    = (float*)d_ws;                         // [4096, 2048] f32
    float*  k_ws    = q_ws + (size_t)BS * 2048;             // [4096, 1024] f32
    float*  attn_ws = k_ws + (size_t)BS * 1024;             // [4096, 2048] f32
    __bf16* qbf     = (__bf16*)(attn_ws + (size_t)BS * 2048);  // [4096, 2048] bf16
    __bf16* kbf     = qbf + (size_t)BS * 2048;              // [4096, 1024] bf16
    __bf16* vbf     = kbf + (size_t)BS * 1024;              // [4096, 1024] bf16

    // QKV projections (V straight to bf16; Q/K stay f32 for RMSNorm accuracy)
    gemm_bf16_wmma_kernel<float><<<dim3(2048 / 64, BS / 64), 256, 0, stream>>>(
        x, Wq, q_ws, 2048, EE);
    gemm_bf16_wmma_kernel<float><<<dim3(1024 / 64, BS / 64), 256, 0, stream>>>(
        x, Wk, k_ws, 1024, EE);
    gemm_bf16_wmma_kernel<__bf16><<<dim3(1024 / 64, BS / 64), 256, 0, stream>>>(
        x, Wv, vbf, 1024, EE);

    // Per-head RMSNorm + partial RoPE; write bf16 activations for attention.
    rmsnorm_rope_kernel<<<dim3(BS * NH), 256, 0, stream>>>(q_ws, qbf, q_scale, NH, ATT_SCALE);
    rmsnorm_rope_kernel<<<dim3(BS * NKV), 256, 0, stream>>>(k_ws, kbf, k_scale, NKV, 1.0f);

    // Windowed softcapped flash attention (bf16 in, fp32 out)
    flash_attn_kernel<<<dim3(SS / 128, NH, BB), 256, FLASH_LDS_BYTES, stream>>>(
        qbf, kbf, vbf, attn_ws);

    // Output projection
    gemm_bf16_wmma_kernel<float><<<dim3(EE / 64, BS / 64), 256, 0, stream>>>(
        attn_ws, Wo, (float*)d_out, EE, NH * HD);
}
